// RiskLSTM_9199819948620
// MI455X (gfx1250) — matrix operand order repaired and checked by reading, not compiled, not run on hardware
//
#include <hip/hip_runtime.h>

#define B_   128
#define S_   512
#define D_   512
#define H_   256
#define G4H  1024
#define NWG  8

typedef __attribute__((ext_vector_type(16))) __bf16 v16bf;
typedef __attribute__((ext_vector_type(8)))  float  v8f;

union Frag {
    v16bf v;
    uint4 q[2];
    unsigned short u[16];
    __device__ Frag() {}
};

__device__ __forceinline__ unsigned short f2bf(float f) {
    unsigned u = __builtin_bit_cast(unsigned, f);
    u += 0x7FFFu + ((u >> 16) & 1u);               // round-to-nearest-even
    return (unsigned short)(u >> 16);
}
__device__ __forceinline__ float bf2f(unsigned short h) {
    unsigned u = ((unsigned)h) << 16;
    return __builtin_bit_cast(float, u);
}
__device__ __forceinline__ unsigned pk2(float lo, float hi2) {
    return (unsigned)f2bf(lo) | ((unsigned)f2bf(hi2) << 16);
}
__device__ __forceinline__ float sigm(float x)  { return 1.f / (1.f + __expf(-x)); }
__device__ __forceinline__ float tanh_(float x) { return 2.f / (1.f + __expf(-2.f * x)) - 1.f; }

// ---------------------------------------------------------------------------
// Cast kernels: one streaming pass so the GEMM hot loop is pure b128->WMMA.
// cast_x: x[b,s,d] f32 -> xbf[m=s*128+b, d] bf16 (time-major for the scan)
// cast_w: W_ih[g,d] f32 -> wbf[g,d] bf16
// ---------------------------------------------------------------------------
__global__ void __launch_bounds__(256)
cast_x_kernel(const float* __restrict__ x, unsigned short* __restrict__ xbf)
{
    long c  = (long)blockIdx.x * 256 + threadIdx.x;   // chunk id, 8 elems each
    int  d0 = (int)(c & 63) * 8;                      // 512/8 = 64 chunks per row
    int  m  = (int)(c >> 6);                          // m = s*128 + b
    int  bb = m & 127, ss = m >> 7;
    const float* p = x + ((long)bb * S_ + ss) * D_ + d0;
    float4 f0 = *(const float4*)(p);
    float4 f1 = *(const float4*)(p + 4);
    uint4 o;
    o.x = pk2(f0.x, f0.y); o.y = pk2(f0.z, f0.w);
    o.z = pk2(f1.x, f1.y); o.w = pk2(f1.z, f1.w);
    *(uint4*)(xbf + (long)m * D_ + d0) = o;
}

__global__ void __launch_bounds__(256)
cast_w_kernel(const float* __restrict__ w, unsigned short* __restrict__ wbf)
{
    long c = (long)blockIdx.x * 256 + threadIdx.x;
    long e = c * 8;
    const float* p = w + e;
    float4 f0 = *(const float4*)(p);
    float4 f1 = *(const float4*)(p + 4);
    uint4 o;
    o.x = pk2(f0.x, f0.y); o.y = pk2(f0.z, f0.w);
    o.z = pk2(f1.x, f1.y); o.w = pk2(f1.z, f1.w);
    *(uint4*)(wbf + e) = o;
}

// ---------------------------------------------------------------------------
// Kernel 1: xg[m][g] = sum_d xbf[m,d]*wbf[g,d] + (b_ih[g]+b_hh[g])
// GEMM M=65536, N=1024, K=512, bf16 WMMA, f32 accumulate.
// Block: 256 thr = 8 waves; wave tile 32x64 (2 M-tiles x 4 N-tiles).
// ---------------------------------------------------------------------------
__global__ void __launch_bounds__(256)
xproj_wmma(const unsigned short* __restrict__ xbf,
           const unsigned short* __restrict__ wbf,
           const float* __restrict__ bih, const float* __restrict__ bhh,
           float* __restrict__ xg)
{
    const int tid  = threadIdx.x;
    const int w    = tid >> 5;
    const int lane = tid & 31;
    const int l15  = lane & 15;
    const int hi   = lane >> 4;

    const int mW = blockIdx.y * 64  + (w & 1) * 32;
    const int nW = blockIdx.x * 256 + (w >> 1) * 64;

    v8f acc[2][4] = {};

    const unsigned short* arow[2];
#pragma unroll
    for (int mt = 0; mt < 2; mt++)
        arow[mt] = xbf + (long)(mW + mt * 16 + l15) * D_ + hi * 8;
    const unsigned short* brow[4];
#pragma unroll
    for (int nt = 0; nt < 4; nt++)
        brow[nt] = wbf + (long)(nW + nt * 16 + l15) * D_ + hi * 8;

    for (int k = 0; k < D_; k += 32) {
        Frag a[2];
#pragma unroll
        for (int mt = 0; mt < 2; mt++) {
            // A frag: K chunks at k+8*hi and k+8*hi+16 (ISA 16-bit A layout)
            const unsigned short* p = arow[mt] + k;
            a[mt].q[0] = *(const uint4*)(p);
            a[mt].q[1] = *(const uint4*)(p + 16);
        }
#pragma unroll
        for (int nt = 0; nt < 4; nt++) {
            // B frag: K chunks at k+8*hi and k+8*hi+16 (same order as A)
            const unsigned short* p = brow[nt] + k;
            Frag b;
            b.q[0] = *(const uint4*)(p);
            b.q[1] = *(const uint4*)(p + 16);
#pragma unroll
            for (int mt = 0; mt < 2; mt++)
                acc[mt][nt] = __builtin_amdgcn_wmma_f32_16x16x32_bf16(
                    false, a[mt].v, false, b.v, (short)0, acc[mt][nt], false, false);
        }
    }

#pragma unroll
    for (int nt = 0; nt < 4; nt++) {
        int g = nW + nt * 16 + l15;
        float bias = bih[g] + bhh[g];
#pragma unroll
        for (int mt = 0; mt < 2; mt++) {
#pragma unroll
            for (int r = 0; r < 8; r++) {
                int m = mW + mt * 16 + r + 8 * hi;   // C/D layout: row = r + 8*(lane/16)
                xg[(long)m * G4H + g] = acc[mt][nt][r] + bias;
            }
        }
    }
}

// ---------------------------------------------------------------------------
// Kernel 2: persistent LSTM recurrence. NWG=8 workgroups, each owns 32
// h-columns. W_hh slice pinned in LDS (64 KB bf16). Per step:
//   1) issue this step's xg loads (independent of h -> overlap the GEMM)
//   2) prefetch next step's xg slice (global_prefetch_b8, L2 warm-up)
//   3) gates = h @ W_hh^T via WMMA (A from L2-resident h, B from LDS)
//   4) gates += xg, LSTM elementwise, store h slice, grid barrier
// ---------------------------------------------------------------------------
__global__ void __launch_bounds__(256)
lstm_rec(const float* __restrict__ xg, const float* __restrict__ Whh,
         unsigned short* __restrict__ hb0, unsigned short* __restrict__ hb1,
         unsigned* __restrict__ cnt)
{
    __shared__ __align__(16) unsigned short ldsW[128 * H_];   // [col][k], 64 KB

    const int tid  = threadIdx.x;
    const int w    = tid >> 5;
    const int lane = tid & 31;
    const int l15  = lane & 15;
    const int hi   = lane >> 4;
    const int j0   = blockIdx.x * 32;   // h-column slice
    const int m0   = w * 16;            // wave's M tile (B rows)

    // Stage W_hh slice: col = q*32+jj maps to gate row g = q*256 + j0 + jj
    for (int e = tid; e < 128 * H_; e += 256) {
        int col = e >> 8, kk = e & 255;
        int q = col >> 5, jj = col & 31;
        ldsW[e] = f2bf(Whh[(q * H_ + j0 + jj) * H_ + kk]);
    }
    __syncthreads();

    int gcol[8];
#pragma unroll
    for (int nt = 0; nt < 8; nt++)
        gcol[nt] = (nt >> 1) * H_ + j0 + (nt & 1) * 16 + l15;
    int mrow[8];
#pragma unroll
    for (int r = 0; r < 8; r++) mrow[r] = m0 + r + 8 * hi;

    float creg[2][8] = {};              // lane-private cell state
    unsigned short* hb[2] = { hb0, hb1 };

    for (int s = 0; s < S_; s++) {
        const unsigned short* hr = hb[s & 1];
        unsigned short*       hw = hb[(s + 1) & 1];

        // (1) issue xg loads first: they only depend on s, so their HBM
        // latency hides under the whole GEMM below (in-order issue).
        const float* xgs = xg + (long)s * (B_ * G4H);
        float xgv[8][8];
#pragma unroll
        for (int nt = 0; nt < 8; nt++) {
            const float* colp = xgs + gcol[nt];
#pragma unroll
            for (int r = 0; r < 8; r++)
                xgv[nt][r] = colp[(long)mrow[r] * G4H];
        }

        // (2) prefetch next step's slice (one per 128B-aligned segment:
        // even nt covers both 16-col halves of each gate q-group).
        int sn = (s + 1 < S_) ? (s + 1) : s;
        const float* xgn = xg + (long)sn * (B_ * G4H);
#pragma unroll
        for (int nt = 0; nt < 8; nt += 2) {
#pragma unroll
            for (int r = 0; r < 8; r++)
                __builtin_prefetch(xgn + gcol[nt] + (long)mrow[r] * G4H, 0, 3);
        }

        // (3) gates = h @ W_hh^T
        v8f acc[8] = {};
        const unsigned short* arowp = hr + (m0 + l15) * H_;
        for (int kk = 0; kk < H_; kk += 32) {
            Frag a;
            a.q[0] = *(const uint4*)(arowp + kk + hi * 16);
            a.q[1] = *(const uint4*)(arowp + kk + hi * 16 + 8);
            int kb = kk + hi * 16;
#pragma unroll
            for (int nt = 0; nt < 8; nt++) {
                Frag b;
                const unsigned short* lp = &ldsW[(nt * 16 + l15) * H_ + kb];
                b.q[0] = *(const uint4*)(lp);
                b.q[1] = *(const uint4*)(lp + 8);
                acc[nt] = __builtin_amdgcn_wmma_f32_16x16x32_bf16(
                    false, a.v, false, b.v, (short)0, acc[nt], false, false);
            }
        }

        // (4) add input projection, LSTM gate math (nt = 2q+t; i,f,g,o)
#pragma unroll
        for (int nt = 0; nt < 8; nt++)
#pragma unroll
            for (int r = 0; r < 8; r++)
                acc[nt][r] += xgv[nt][r];

#pragma unroll
        for (int t = 0; t < 2; t++) {
#pragma unroll
            for (int r = 0; r < 8; r++) {
                float ig = sigm (acc[0 + t][r]);
                float fg = sigm (acc[2 + t][r]);
                float gg = tanh_(acc[4 + t][r]);
                float og = sigm (acc[6 + t][r]);
                float c  = fg * creg[t][r] + ig * gg;
                creg[t][r] = c;
                float h  = og * tanh_(c);
                hw[(long)mrow[r] * H_ + (j0 + 16 * t + l15)] = f2bf(h);
            }
        }

        // grid-wide barrier (monotonic counter; 8 resident blocks)
        __threadfence();
        __syncthreads();
        if (tid == 0) {
            __hip_atomic_fetch_add(cnt, 1u, __ATOMIC_RELEASE, __HIP_MEMORY_SCOPE_AGENT);
            unsigned target = (unsigned)(NWG * (s + 1));
            while (__hip_atomic_load(cnt, __ATOMIC_ACQUIRE, __HIP_MEMORY_SCOPE_AGENT) < target)
                __builtin_amdgcn_s_sleep(1);
        }
        __syncthreads();
    }
}

// ---------------------------------------------------------------------------
// Kernel 3: logits = h_n @ fc_w^T + fc_b   ([128,256] x [256,1])
// ---------------------------------------------------------------------------
__global__ void fc_kernel(const unsigned short* __restrict__ hb,
                          const float* __restrict__ fw,
                          const float* __restrict__ fb,
                          float* __restrict__ out)
{
    int m = threadIdx.x;
    float s = 0.f;
    for (int k = 0; k < H_; k++)
        s += bf2f(hb[m * H_ + k]) * fw[k];
    out[m] = s + fb[0];
}

extern "C" void kernel_launch(void* const* d_in, const int* in_sizes, int n_in,
                              void* d_out, int out_size, void* d_ws, size_t ws_size,
                              hipStream_t stream)
{
    (void)in_sizes; (void)n_in; (void)out_size; (void)ws_size;

    const float* x   = (const float*)d_in[0];
    const float* Wih = (const float*)d_in[1];
    const float* Whh = (const float*)d_in[2];
    const float* bih = (const float*)d_in[3];
    const float* bhh = (const float*)d_in[4];
    const float* fw  = (const float*)d_in[5];
    const float* fb  = (const float*)d_in[6];
    float* out = (float*)d_out;

    char* ws = (char*)d_ws;
    const size_t XG_BYTES  = (size_t)S_ * B_ * G4H * sizeof(float);          // 256 MB
    const size_t XBF_BYTES = (size_t)S_ * B_ * D_ * sizeof(unsigned short);  // 64 MB
    const size_t WBF_BYTES = (size_t)G4H * D_ * sizeof(unsigned short);      // 1 MB
    const size_t HB_BYTES  = (size_t)B_ * H_ * sizeof(unsigned short);       // 64 KB

    float*          xgbuf = (float*)ws;
    unsigned short* xbf   = (unsigned short*)(ws + XG_BYTES);
    unsigned short* wbf   = (unsigned short*)(ws + XG_BYTES + XBF_BYTES);
    unsigned short* hb0   = (unsigned short*)(ws + XG_BYTES + XBF_BYTES + WBF_BYTES);
    unsigned short* hb1   = hb0 + B_ * H_;
    unsigned*       cnt   = (unsigned*)((char*)hb0 + 2 * HB_BYTES);

    // zero h0 buffers + barrier counter each launch (graph-capture safe)
    hipMemsetAsync((void*)hb0, 0, 2 * HB_BYTES + 64, stream);

    // one-pass f32 -> bf16 casts (x also transposed to time-major rows)
    long xchunks = (long)S_ * B_ * D_ / 8;            // 4,194,304
    cast_x_kernel<<<(unsigned)(xchunks / 256), 256, 0, stream>>>(x, xbf);
    long wchunks = (long)G4H * D_ / 8;                // 65,536
    cast_w_kernel<<<(unsigned)(wchunks / 256), 256, 0, stream>>>(Wih, wbf);

    dim3 gproj(G4H / 256, (S_ * B_) / 64);
    xproj_wmma<<<gproj, 256, 0, stream>>>(xbf, wbf, bih, bhh, xgbuf);
    lstm_rec<<<NWG, 256, 0, stream>>>(xgbuf, Whh, hb0, hb1, cnt);
    fc_kernel<<<1, B_, 0, stream>>>(hb0, fw, fb, out);
}
